// OPTAttention_63496796504656
// MI455X (gfx1250) — matrix operand order repaired and checked by reading, not compiled, not run on hardware
//
#include <hip/hip_runtime.h>
#include <hip/hip_bf16.h>

// ---------------------------------------------------------------------------
// OPT attention block (B=2, S=2048, E=2048, H=32, D=64) for gfx1250 / MI455X.
// All matmuls via v_wmma_f32_16x16x32_f16 (f16 in, f32 accumulate).
//  - Projections: 16Mx128N wave tiles, 8 WMMAs / 32-deep k-step, 2-stage
//    ping-pong pipeline (no buffer-rotation copies)
//  - V projection writes V^T per head ([B,H,D,S]) so flash-attention P*V
//    B-fragments are contiguous global loads
//  - Flash attention ping-pongs K/V fragment sets across key blocks; softmax
//    row-sums computed by WMMA against a ones matrix
// ---------------------------------------------------------------------------

typedef _Float16 h16;
typedef __attribute__((ext_vector_type(16))) _Float16 v16h;
typedef __attribute__((ext_vector_type(8)))  _Float16 v8h;
typedef __attribute__((ext_vector_type(8)))  float    v8f;

static constexpr int B_  = 2;
static constexpr int S_  = 2048;
static constexpr int E_  = 2048;
static constexpr int H_  = 32;
static constexpr int D_  = 64;
static constexpr float SCALE_  = 0.125f;   // 64^-0.5
static constexpr float NEGINF_ = -1e30f;

#define WMMA_F16(a, b, c) \
  __builtin_amdgcn_wmma_f32_16x16x32_f16(false, (a), false, (b), (short)0, (c), false, false)

#define CAT16(lo, hi) \
  __builtin_shufflevector((lo), (hi), 0, 1, 2, 3, 4, 5, 6, 7, 8, 9, 10, 11, 12, 13, 14, 15)

// ---------------------------------------------------------------------------
// fp32 -> f16 conversion (grid-stride)
// ---------------------------------------------------------------------------
__global__ void f32_to_f16_kernel(const float* __restrict__ src,
                                  h16* __restrict__ dst, size_t n) {
  size_t i = (size_t)blockIdx.x * blockDim.x + threadIdx.x;
  size_t stride = (size_t)gridDim.x * blockDim.x;
  for (; i < n; i += stride) dst[i] = (h16)src[i];
}

// ---------------------------------------------------------------------------
// C[m,n] = (sum_k X[m,k] * W[n,k] + bias[n]) * scale     (i.e. X @ W.T + b)
// Wave tile: 16M x 128N. Block: 8 waves = 32M x 512N.
// Ping-pong pipeline over k (K/32 steps is even): stage loads alternate
// between two fragment sets -> no rotation copies.
// OUT_MODE: 0 = f16 [M,N], 1 = f32 [M,N], 2 = f16 V^T per head [B,H,D,S]
// ---------------------------------------------------------------------------
template <int OUT_MODE>
__global__ __launch_bounds__(256)
void gemm_xwT_kernel(const h16* __restrict__ X, const h16* __restrict__ W,
                     const float* __restrict__ bias, void* __restrict__ outp,
                     int M, int N, int K, float scale) {
  const int lane = threadIdx.x & 31;
  const int wave = threadIdx.x >> 5;
  const int wm = wave >> 2, wn = wave & 3;
  const int m0 = blockIdx.x * 32 + wm * 16;
  const int n0 = blockIdx.y * 512 + wn * 128;
  const int lm = lane & 15;
  const int g  = lane >> 4;            // lane half

  // A-frag base: lane row = m0+lm, k-chunks {kb..kb+7, kb+8..kb+15}, kb=16g
  const h16* abase = X + (size_t)(m0 + lm) * K + (g ? 16 : 0);
  // B-frag base: lane col = n0+lm, K 0..15 / 16..31 per half; nb adds 16*K
  const h16* bbase = W + (size_t)(n0 + lm) * K + (g ? 16 : 0);

  float bb[8];
#pragma unroll
  for (int nb = 0; nb < 8; ++nb) bb[nb] = bias[n0 + nb * 16 + lm];

  // ---- prologue: stage k=0 into set 0 ----
  v8h a0lo = *(const v8h*)(abase);
  v8h a0hi = *(const v8h*)(abase + 8);
  v16h b0[8];
#pragma unroll
  for (int nb = 0; nb < 8; ++nb)
    b0[nb] = *(const v16h*)(bbase + (size_t)nb * 16 * K);

  v8f acc[8] = {};
  for (int k = 0; k < K; k += 64) {
    // ---- stage A: load k+32 into set 1, compute with set 0 ----
    const v8h a1lo = *(const v8h*)(abase + k + 32);
    const v8h a1hi = *(const v8h*)(abase + k + 40);
    v16h b1[8];
#pragma unroll
    for (int nb = 0; nb < 8; ++nb)
      b1[nb] = *(const v16h*)(bbase + (size_t)nb * 16 * K + k + 32);
    {
      const v16h a = CAT16(a0lo, a0hi);
#pragma unroll
      for (int nb = 0; nb < 8; ++nb)
        acc[nb] = WMMA_F16(a, b0[nb], acc[nb]);
    }
    // ---- stage B: load k+64 (clamped) into set 0, compute with set 1 ----
    const int k2 = (k + 64 < K) ? (k + 64) : 0;   // harmless reload at end
    a0lo = *(const v8h*)(abase + k2);
    a0hi = *(const v8h*)(abase + k2 + 8);
#pragma unroll
    for (int nb = 0; nb < 8; ++nb)
      b0[nb] = *(const v16h*)(bbase + (size_t)nb * 16 * K + k2);
    {
      const v16h a = CAT16(a1lo, a1hi);
#pragma unroll
      for (int nb = 0; nb < 8; ++nb)
        acc[nb] = WMMA_F16(a, b1[nb], acc[nb]);
    }
  }

  // C/D layout: element r of lane = row m0 + r + 8*g, col n0 + nb*16 + lm
#pragma unroll
  for (int nb = 0; nb < 8; ++nb) {
    const int col = n0 + nb * 16 + lm;
    if constexpr (OUT_MODE == 2) {
      // V^T per head: Vt[((b*H + h)*D + d)*S + s], s = row - b*S
      const int hh = col >> 6, dd = col & 63;
      const int bb2 = m0 >> 11;            // m0 multiple of 16: whole tile same b
      const int ss0 = (m0 & (S_ - 1)) + 8 * g;
      v8h vals;
#pragma unroll
      for (int r = 0; r < 8; ++r) vals[r] = (h16)((acc[nb][r] + bb[nb]) * scale);
      h16* dst = (h16*)outp + (((size_t)bb2 * H_ + hh) * D_ + dd) * S_ + ss0;
      *(v8h*)dst = vals;                    // 8 consecutive s values
    } else {
#pragma unroll
      for (int r = 0; r < 8; ++r) {
        const int row = m0 + r + 8 * g;
        const float v = (acc[nb][r] + bb[nb]) * scale;
        if constexpr (OUT_MODE == 0) ((h16*)outp)[(size_t)row * N + col] = (h16)v;
        else                         ((float*)outp)[(size_t)row * N + col] = v;
      }
    }
  }
}

// ---------------------------------------------------------------------------
// Causal flash attention: one wave (32 threads) per (b, h, 16-row q-tile).
// K/V fragment sets ping-pong across key blocks (no rotation copies);
// next block's loads issue right after the score WMMAs and fly under the
// softmax VALU section + LDS round-trip.
// ---------------------------------------------------------------------------
__global__ __launch_bounds__(32)
void attn_kernel(const h16* __restrict__ Q, const h16* __restrict__ Kh,
                 const h16* __restrict__ Vt, h16* __restrict__ O) {
  __shared__ alignas(16) h16 lds_p[16][40];   // 16x32 P tile, padded rows

  const int lane = threadIdx.x;
  const int lm = lane & 15;
  const int g  = lane >> 4;

  int bid = blockIdx.x;
  const int QT = S_ / 16;
  const int qt = bid % QT; bid /= QT;
  const int h  = bid % H_; bid /= H_;
  const int b  = bid;
  const int q0 = qt * 16;

  const size_t headoff = (size_t)h * D_;

  // Q A-frags for d-chunks [0,32) and [32,64)
  const h16* qrow = Q + ((size_t)b * S_ + (q0 + lm)) * E_ + headoff;
  const int kba = g ? 16 : 0;
  const v16h qa0 = CAT16(*(const v8h*)(qrow + kba),      *(const v8h*)(qrow + 8 + kba));
  const v16h qa1 = CAT16(*(const v8h*)(qrow + 32 + kba), *(const v8h*)(qrow + 40 + kba));

  // ones B-matrix for rowsum WMMA
  v16h onesb;
#pragma unroll
  for (int i = 0; i < 16; ++i) onesb[i] = (h16)1.0f;

  // base pointers
  const h16* kbase  = Kh + ((size_t)b * S_ + lm) * E_ + headoff + (g ? 16 : 0);
  const h16* vtbase = Vt + (((size_t)b * H_ + h) * D_ + lm) * S_ + (g ? 16 : 0);

  float m_[8], l_[8];
#pragma unroll
  for (int r = 0; r < 8; ++r) { m_[r] = NEGINF_; l_[r] = 0.0f; }
  v8f acc[4] = {};

  auto loadKV = [&](int n, v16h* kf, v16h* vf) {
    const h16* kr = kbase + (size_t)n * E_;
    kf[0] = *(const v16h*)(kr);
    kf[1] = *(const v16h*)(kr + 32);
    kf[2] = *(const v16h*)(kr + (size_t)16 * E_);
    kf[3] = *(const v16h*)(kr + (size_t)16 * E_ + 32);
#pragma unroll
    for (int nb = 0; nb < 4; ++nb)
      vf[nb] = *(const v16h*)(vtbase + (size_t)nb * 16 * S_ + n);
  };

  // process block j using (kc, vc); prefetch block j+1 into (kn, vn)
  auto processBlock = [&](int j, v16h* kc, v16h* vc, v16h* kn, v16h* vn) {
    const int n0 = j * 32;

    // ---- score tiles with current K frags ----
    v8f s0 = {}, s1 = {};
    s0 = WMMA_F16(qa0, kc[0], s0);
    s0 = WMMA_F16(qa1, kc[1], s0);
    s1 = WMMA_F16(qa0, kc[2], s1);
    s1 = WMMA_F16(qa1, kc[3], s1);

    // ---- issue next block's loads (hide under softmax VALU work) ----
    const int n1 = (n0 + 32 <= S_ - 32) ? (n0 + 32) : 0;  // clamp: harmless
    loadKV(n1, kn, vn);

    // ---- causal mask (diagonal blocks only) ----
    if (n0 + 31 > q0) {
#pragma unroll
      for (int r = 0; r < 8; ++r) {
        const int q = q0 + r + 8 * g;
        if (n0 + lm      > q) s0[r] = NEGINF_;
        if (n0 + 16 + lm > q) s1[r] = NEGINF_;
      }
    }

    // ---- online softmax: row max via shfl butterfly over 16-lane half ----
    float alpha[8];
#pragma unroll
    for (int r = 0; r < 8; ++r) {
      float t = fmaxf(s0[r], s1[r]);
      t = fmaxf(t, __shfl_xor(t, 1, 32));
      t = fmaxf(t, __shfl_xor(t, 2, 32));
      t = fmaxf(t, __shfl_xor(t, 4, 32));
      t = fmaxf(t, __shfl_xor(t, 8, 32));
      const float mn = fmaxf(m_[r], t);
      alpha[r] = __expf(m_[r] - mn);
      m_[r] = mn;
      // masked entries underflow to 0 in exp
      lds_p[r + 8 * g][lm]      = (h16)__expf(s0[r] - mn);
      lds_p[r + 8 * g][16 + lm] = (h16)__expf(s1[r] - mn);
    }
#pragma unroll
    for (int nb = 0; nb < 4; ++nb)
#pragma unroll
      for (int r = 0; r < 8; ++r) acc[nb][r] *= alpha[r];

    __syncthreads();   // single-wave WG: compiler fence; barrier -> s_nop

    // ---- P A-frag (16x32) from LDS ----
    const v16h pa = CAT16(*(const v8h*)(&lds_p[lm][kba]),
                          *(const v8h*)(&lds_p[lm][8 + kba]));

    // ---- row sums via WMMA against ones matrix (same C/D layout as l_) ----
    {
      v8f rs = {};
      rs = WMMA_F16(pa, onesb, rs);
#pragma unroll
      for (int r = 0; r < 8; ++r) l_[r] = l_[r] * alpha[r] + rs[r];
    }

    // ---- P(16x32) x V(32x64) with current V frags ----
#pragma unroll
    for (int nb = 0; nb < 4; ++nb)
      acc[nb] = WMMA_F16(pa, vc[nb], acc[nb]);

    __syncthreads();
  };

  const int jmax = (q0 + 15) >> 5;
  v16h k0f[4], v0f[4], k1f[4], v1f[4];
  loadKV(0, k0f, v0f);

  int j = 0;
  while (true) {
    processBlock(j, k0f, v0f, k1f, v1f);   // uses set0, prefetches into set1
    if (++j > jmax) break;
    processBlock(j, k1f, v1f, k0f, v0f);   // uses set1, prefetches into set0
    if (++j > jmax) break;
  }

  // ---- epilogue: O[q, h*64 + d] = acc / l ----
#pragma unroll
  for (int r = 0; r < 8; ++r) {
    const float inv = 1.0f / l_[r];
    const int q = q0 + r + 8 * g;
    h16* op = O + ((size_t)b * S_ + q) * E_ + headoff;
#pragma unroll
    for (int nb = 0; nb < 4; ++nb)
      op[nb * 16 + lm] = (h16)(acc[nb][r] * inv);
  }
}

// ---------------------------------------------------------------------------
// launch
// ---------------------------------------------------------------------------
extern "C" void kernel_launch(void* const* d_in, const int* in_sizes, int n_in,
                              void* d_out, int out_size, void* d_ws, size_t ws_size,
                              hipStream_t stream) {
  const float* hs = (const float*)d_in[0];
  const float* Wq = (const float*)d_in[1];
  const float* bq = (const float*)d_in[2];
  const float* Wk = (const float*)d_in[3];
  const float* bk = (const float*)d_in[4];
  const float* Wv = (const float*)d_in[5];
  const float* bv = (const float*)d_in[6];
  const float* Wo = (const float*)d_in[7];
  const float* bo = (const float*)d_in[8];

  const size_t XE = (size_t)B_ * S_ * E_;   // 8.4M elements
  const size_t WE = (size_t)E_ * E_;        // 4.2M elements

  h16* p = (h16*)d_ws;                      // ~112 MB of f16 scratch
  h16* Xh   = p; p += XE;
  h16* Wq16 = p; p += WE;
  h16* Wk16 = p; p += WE;
  h16* Wv16 = p; p += WE;
  h16* Wo16 = p; p += WE;
  h16* Qh   = p; p += XE;
  h16* Kh   = p; p += XE;
  h16* Vth  = p; p += XE;                   // V^T, [B,H,D,S]
  h16* Ah   = p; p += XE;

  f32_to_f16_kernel<<<2048, 256, 0, stream>>>(hs, Xh, XE);
  f32_to_f16_kernel<<<1024, 256, 0, stream>>>(Wq, Wq16, WE);
  f32_to_f16_kernel<<<1024, 256, 0, stream>>>(Wk, Wk16, WE);
  f32_to_f16_kernel<<<1024, 256, 0, stream>>>(Wv, Wv16, WE);
  f32_to_f16_kernel<<<1024, 256, 0, stream>>>(Wo, Wo16, WE);

  const int M = B_ * S_;                    // 4096
  dim3 gg(M / 32, E_ / 512);                // (128, 4)
  // SCALE folded into Q (bias included: (x@Wq.T + bq) * SCALE)
  gemm_xwT_kernel<0><<<gg, 256, 0, stream>>>(Xh, Wq16, bq, Qh,  M, E_, E_, SCALE_);
  gemm_xwT_kernel<0><<<gg, 256, 0, stream>>>(Xh, Wk16, bk, Kh,  M, E_, E_, 1.0f);
  gemm_xwT_kernel<2><<<gg, 256, 0, stream>>>(Xh, Wv16, bv, Vth, M, E_, E_, 1.0f);

  attn_kernel<<<B_ * H_ * (S_ / 16), 32, 0, stream>>>(Qh, Kh, Vth, Ah);

  gemm_xwT_kernel<1><<<gg, 256, 0, stream>>>(Ah, Wo16, bo, d_out, M, E_, E_, 1.0f);
}
